// NEMModel_27504970563844
// MI455X (gfx1250) — matrix operand order repaired and checked by reading, not compiled, not run on hardware
//
#include <hip/hip_runtime.h>

// ---------------------------------------------------------------------------
// Types for CDNA5 WMMA (wave32) + TDM descriptor groups
// ---------------------------------------------------------------------------
typedef __attribute__((ext_vector_type(16))) _Float16 v16h;
typedef __attribute__((ext_vector_type(8)))  float    v8f;
typedef __attribute__((ext_vector_type(4)))  unsigned int u32x4;
typedef __attribute__((ext_vector_type(8)))  int      i32x8;
typedef __attribute__((ext_vector_type(4)))  int      i32x4;

#define D 256          // D_H == D_A
#define GAMMA 0.001f

#if defined(__has_builtin)
#if __has_builtin(__builtin_amdgcn_tensor_load_to_lds)
#define USE_TDM 1
#endif
#endif
#ifndef USE_TDM
#define USE_TDM 0
#endif

// ---------------------------------------------------------------------------
// Weight pre-swizzle: pack one 256-column window of W (row-major f32,
// row stride wst, column offset folded into W) into f16 WMMA B-fragments:
//   out[((ks*16 + nb)*32 + lane)*16 + t] = W[(nb*16 + lane%16)*wst + ks*32 + (lane/16)*16 + t]
// ---------------------------------------------------------------------------
__global__ __launch_bounds__(256) void wswz_k(const float* __restrict__ W, int wst,
                                              _Float16* __restrict__ out)
{
    const int idx  = blockIdx.x * 256 + threadIdx.x;
    const int lane = idx & 31;
    const int frag = idx >> 5;          // 0..127
    const int ks   = frag >> 4;
    const int nb   = frag & 15;
    const int n    = nb * 16 + (lane & 15);
    const int kb   = ks * 32 + (lane >> 4) * 16;
    const float* __restrict__ wp = W + (size_t)n * wst + kb;
    _Float16* __restrict__ op = out + (size_t)idx * 16;
#pragma unroll
    for (int t = 0; t < 16; ++t) op[t] = (_Float16)wp[t];
}

// ---------------------------------------------------------------------------
// Stage one 128 KB pre-swizzled B window (global) into LDS.
// TDM path: one wave issues a single 2D tensor DMA (8192 x 8 2-byte elems),
// waits on TENSORcnt. Fallback: cooperative float4 copy loop.
// Caller must __syncthreads() after (and before, if LDS is being reused).
// ---------------------------------------------------------------------------
__device__ __forceinline__ void stage_b_window(const _Float16* __restrict__ Bs,
                                               _Float16* ldsB)
{
#if USE_TDM
    if (threadIdx.x < 32) {           // one wave issues the DMA
        typedef __attribute__((address_space(3))) void lds_void;
        const unsigned lds_off = (unsigned)(size_t)(lds_void*)ldsB;
        const unsigned long long ga = (unsigned long long)(size_t)Bs;
        // ---- D# group 0 ----
        u32x4 g0;
        g0[0] = 1u;                                              // count = 1
        g0[1] = lds_off;                                         // lds_addr (bytes)
        g0[2] = (unsigned)(ga & 0xFFFFFFFFull);                  // global_addr[31:0]
        g0[3] = (unsigned)((ga >> 32) & 0x01FFFFFFull)           // global_addr[56:32]
              | (2u << 30);                                      // type = 2 (image)
        // ---- D# group 1 ----  data_size=2B, tensor/tile = 8192 x 8, stride 8192
        i32x8 g1;
        g1[0] = 0x00010000;   // data_size = 1 (2 bytes) at bits 17:16
        g1[1] = 0x20000000;   // tensor_dim0 = 8192 (bits 79:48, low half)
        g1[2] = 0x00080000;   // tensor_dim1 = 8 (bits 111:80, low half)
        g1[3] = 0x20000000;   // tile_dim0 = 8192 (bits 127:112)
        g1[4] = 8;            // tile_dim1 = 8 (bits 143:128)
        g1[5] = 8192;         // tensor_dim0_stride = 8192 (bits 207:160, low)
        g1[6] = 0x20000000;   // tensor_dim1_stride = 8192 (unused for 2D tile)
        g1[7] = 0;
        const i32x4 z4 = (i32x4)(0);
        const i32x8 z8 = (i32x8)(0);
        __builtin_amdgcn_tensor_load_to_lds(g0, g1, z4, z4, z8, 0);
        __builtin_amdgcn_s_wait_tensorcnt(0);
    }
#else
    float4* __restrict__ d = (float4*)ldsB;
    const float4* __restrict__ s = (const float4*)Bs;
    for (int i = threadIdx.x; i < 8192; i += 256) d[i] = s[i];
#endif
}

// ---------------------------------------------------------------------------
// Fused GEMM:  C = act( concat(X0[,X1[,X2]]) @ W.T + bias0 [+ bias1] )
//   Xs: (M x 256) f32 row-major. Bs: pre-swizzled f16 B-fragments (wswz_k),
//   staged per-source into LDS (128 KB) and consumed via ds_load_b128.
//   ACT: 0=none 1=leaky_relu(0.01) 2=tanh 3=row l2-norm (after bias)
//   One wave computes 16 rows x 256 cols: A loaded once, 16 WMMA acc tiles.
//   Block = 256 threads = 8 waves = 128 rows. grid.x = ceil(M/128).
//   Launch with 131072 bytes of dynamic LDS.
// ---------------------------------------------------------------------------
template<int NSRC, int ACT>
__global__ __launch_bounds__(256) void gemm_wmma_k(
    const float* __restrict__ X0, const float* __restrict__ X1, const float* __restrict__ X2,
    const _Float16* __restrict__ B0, const _Float16* __restrict__ B1,
    const _Float16* __restrict__ B2,
    const float* __restrict__ bias0, const float* __restrict__ bias1,
    float* __restrict__ C, int M)
{
    extern __shared__ _Float16 ldsB[];          // 65536 f16 = 128 KB window

    const int lane  = threadIdx.x & 31;
    const int wave  = threadIdx.x >> 5;
    const int tileM = (blockIdx.x * 8 + wave) * 16;
    const int half  = lane >> 4;      // 0: lanes 0-15, 1: lanes 16-31
    const int l15   = lane & 15;

    v8f acc[16];
#pragma unroll
    for (int j = 0; j < 16; ++j) acc[j] = (v8f)(0.0f);

    const int  arow    = tileM + l15;       // A row this lane supplies
    const bool arow_ok = (arow < M);

    auto do_src = [&](const float* __restrict__ X, const _Float16* __restrict__ Bs) {
        __syncthreads();                        // previous window fully consumed
        stage_b_window(Bs, ldsB);
        __syncthreads();                        // window visible to all waves

        const float* __restrict__ ap = X + (size_t)arow * D;
        __builtin_prefetch(ap, 0, 1);           // global_prefetch_b8: warm A row
        for (int ks = 0; ks < 8; ++ks) {
            const int kb = ks * 32;
            // ---- A fragment: 16x32 f16, contiguous k = 16*half + s ----
            v16h afrag;
            {
                const int k0 = kb + half * 16;
                const int k1 = kb + half * 16 + 8;
#pragma unroll
                for (int t = 0; t < 8; ++t) {
                    float f0 = arow_ok ? ap[k0 + t] : 0.0f;
                    float f1 = arow_ok ? ap[k1 + t] : 0.0f;
                    afrag[t]     = (_Float16)f0;
                    afrag[8 + t] = (_Float16)f1;
                }
            }
            // ---- 16 N-subtiles from LDS, one fragment ahead (ds || wmma) ----
            const _Float16* bp = ldsB + ((size_t)(ks * 16) * 32 + lane) * 16;
            v16h bcur = *(const v16h*)(bp);
#pragma unroll
            for (int nb = 0; nb < 16; ++nb) {
                v16h bnext = bcur;
                if (nb < 15) bnext = *(const v16h*)(bp + (size_t)(nb + 1) * 512);
                acc[nb] = __builtin_amdgcn_wmma_f32_16x16x32_f16(
                    false, afrag, false, bcur, (short)0, acc[nb], false, false);
                bcur = bnext;
            }
        }
    };

    do_src(X0, B0);
    if (NSRC > 1) do_src(X1, B1);
    if (NSRC > 2) do_src(X2, B2);

    // ---- epilogue: bias ----
#pragma unroll
    for (int nb = 0; nb < 16; ++nb) {
        float b = bias0[nb * 16 + l15];
        if (bias1) b += bias1[nb * 16 + l15];
#pragma unroll
        for (int r = 0; r < 8; ++r) acc[nb][r] += b;
    }

    if (ACT == 1) {                       // leaky relu 0.01
#pragma unroll
        for (int nb = 0; nb < 16; ++nb)
#pragma unroll
            for (int r = 0; r < 8; ++r) {
                float v = acc[nb][r];
                acc[nb][r] = v > 0.0f ? v : 0.01f * v;
            }
    } else if (ACT == 2) {                // tanh
#pragma unroll
        for (int nb = 0; nb < 16; ++nb)
#pragma unroll
            for (int r = 0; r < 8; ++r) acc[nb][r] = tanhf(acc[nb][r]);
    } else if (ACT == 3) {                // row l2-norm: p / sqrt(sum p^2 + 1e-6)
#pragma unroll
        for (int r = 0; r < 8; ++r) {
            float ss = 0.0f;
#pragma unroll
            for (int nb = 0; nb < 16; ++nb) ss += acc[nb][r] * acc[nb][r];
            ss += __shfl_xor(ss, 1, 16);
            ss += __shfl_xor(ss, 2, 16);
            ss += __shfl_xor(ss, 4, 16);
            ss += __shfl_xor(ss, 8, 16);
            const float sc = rsqrtf(ss + 1e-6f);
#pragma unroll
            for (int nb = 0; nb < 16; ++nb) acc[nb][r] *= sc;
        }
    }

    // ---- store (C layout: VGPR r -> rows tileM+r / tileM+r+8) ----
#pragma unroll
    for (int r = 0; r < 8; ++r) {
        const int row = tileM + r + half * 8;
        if (row < M) {
#pragma unroll
            for (int nb = 0; nb < 16; ++nb)
                C[(size_t)row * D + nb * 16 + l15] = acc[nb][r];
        }
    }
}

// ---------------------------------------------------------------------------
__global__ void outer_k(const float* __restrict__ x, const float* __restrict__ w,
                        const float* __restrict__ b, float* __restrict__ out)
{
    const int row = blockIdx.x, tid = threadIdx.x;
    out[(size_t)row * D + tid] = x[row] * w[tid] + b[tid];
}

__global__ void onehot_k(const int* __restrict__ Y, float* __restrict__ out)
{
    const int row = blockIdx.x, tid = threadIdx.x;
    out[(size_t)row * D + tid] = (row == *Y) ? 1.0f : 0.0f;
}

__global__ void zero_k(float* __restrict__ p, size_t n)
{
    const size_t i = (size_t)blockIdx.x * blockDim.x + threadIdx.x;
    if (i < n) p[i] = 0.0f;
}

// out[row,:] = sum_j vals[row*k+j] * Y[cols[row*k+j], :]   (rows structured)
__global__ void spmm_k(const int* __restrict__ cols, const float* __restrict__ vals,
                       const float* __restrict__ Y, float* __restrict__ out, int k)
{
    const int row = blockIdx.x, tid = threadIdx.x;
    const long base = (long)row * k;
    float acc = 0.0f;
    for (int j = 0; j < k; ++j) {
        const int   c = cols[base + j];
        const float v = vals[base + j];
        acc += v * Y[(size_t)c * D + tid];
    }
    out[(size_t)row * D + tid] = acc;
}

// scatter: out[cols[e], :] += vals[e] * H[rows[e], :]   (one wave per edge)
__global__ __launch_bounds__(256) void spmm_t_k(
    const int* __restrict__ rows, const int* __restrict__ cols,
    const float* __restrict__ vals, const float* __restrict__ H,
    float* __restrict__ out, int nnz)
{
    const int lane = threadIdx.x & 31;
    const int wave = threadIdx.x >> 5;
    const int e = blockIdx.x * 8 + wave;
    if (e >= nnz) return;
    const int r = rows[e], c = cols[e];
    const float v = vals[e];
    const float* __restrict__ hp = H + (size_t)r * D + lane * 8;
    float* __restrict__ op = out + (size_t)c * D + lane * 8;
#pragma unroll
    for (int t = 0; t < 8; ++t)
        unsafeAtomicAdd(op + t, v * hp[t]);
}

// new_vals[e] = vals[e] + gamma * dot(Pprev[cols[e]], Pnext[rows[e]])
__global__ __launch_bounds__(256) void dw_k(
    const int* __restrict__ rows, const int* __restrict__ cols,
    const float* __restrict__ vals,
    const float* __restrict__ Pprev, const float* __restrict__ Pnext,
    float* __restrict__ out, int nnz)
{
    const int lane = threadIdx.x & 31;
    const int wave = threadIdx.x >> 5;
    const int e = blockIdx.x * 8 + wave;
    if (e >= nnz) return;
    const int r = rows[e], c = cols[e];
    const float4* pp = (const float4*)(Pprev + (size_t)c * D) + lane * 2;
    const float4* pn = (const float4*)(Pnext + (size_t)r * D) + lane * 2;
    const float4 a0 = pp[0], a1 = pp[1], b0 = pn[0], b1 = pn[1];
    float d = a0.x * b0.x + a0.y * b0.y + a0.z * b0.z + a0.w * b0.w
            + a1.x * b1.x + a1.y * b1.y + a1.z * b1.z + a1.w * b1.w;
    d += __shfl_xor(d, 1);
    d += __shfl_xor(d, 2);
    d += __shfl_xor(d, 4);
    d += __shfl_xor(d, 8);
    d += __shfl_xor(d, 16);
    if (lane == 0) out[e] = vals[e] + GAMMA * d;
}

// ---------------------------------------------------------------------------
extern "C" void kernel_launch(void* const* d_in, const int* in_sizes, int n_in,
                              void* d_out, int out_size, void* d_ws, size_t ws_size,
                              hipStream_t stream)
{
    (void)in_sizes; (void)n_in; (void)out_size; (void)ws_size;
    const int NIN = 16384, NHID = 65536, NOUT = 1000;
    const int K0 = 9, K1 = 11, K2 = 11;
    const int nnz0 = NHID * K0, nnz1 = NHID * K1, nnz2 = NOUT * K2;
    const size_t LDSB = 131072;                 // dynamic LDS for B window

    // inputs (setup_inputs dict order)
    const int*   idx0 = (const int*)d_in[0];
    const float* val0 = (const float*)d_in[1];
    const float* h0   = (const float*)d_in[2];
    const int*   idx1 = (const int*)d_in[3];
    const float* val1 = (const float*)d_in[4];
    const float* h1   = (const float*)d_in[5];
    const int*   idx2 = (const int*)d_in[6];
    const float* val2 = (const float*)d_in[7];
    const float* h2   = (const float*)d_in[8];
    const float* Wvec = (const float*)d_in[9];
    const float* bvec = (const float*)d_in[10];
    const float* Wfirst = (const float*)d_in[11];
    const float* bfirst = (const float*)d_in[12];
    const float* Wact = (const float*)d_in[13];   // (256, 512)
    const float* bact = (const float*)d_in[14];
    const float* Wun  = (const float*)d_in[15];   // (256, 256)
    const float* bun  = (const float*)d_in[16];
    const float* Wup  = (const float*)d_in[17];
    const float* bup  = (const float*)d_in[18];
    const float* Wih  = (const float*)d_in[19];   // (256, 512)
    const float* bih  = (const float*)d_in[20];
    const float* Whh  = (const float*)d_in[21];   // (256, 256)
    const float* bhh  = (const float*)d_in[22];
    const float* x    = (const float*)d_in[23];
    const int*   Yp   = (const int*)d_in[24];

    const int *rows0 = idx0, *cols0 = idx0 + nnz0;
    const int *rows1 = idx1, *cols1 = idx1 + nnz1;
    const int *rows2 = idx2, *cols2 = idx2 + nnz2;

    // outputs
    const size_t BIG = (size_t)NHID * D;   // 16,777,216
    const size_t MED = (size_t)NIN * D;
    const size_t SM  = (size_t)NOUT * D;
    float* out = (float*)d_out;
    float* nh0 = out;
    float* nh1 = out + BIG;
    float* nh2 = out + 2 * BIG;
    float* nv0 = nh2 + SM;
    float* nv1 = nv0 + nnz0;
    float* nv2 = nv1 + nnz1;

    // workspace layout (~238 MB)
    float* bufA  = (float*)d_ws;        // BIG  m_prev0 -> p_prev1 stage
    float* bufB  = bufA + BIG;          // BIG  a0/a1 -> m_next -> p_prev2
    float* bufC  = bufB + BIG;          // BIG  m_prev1 -> p_next0/p_next1
    float* bufS0 = bufC + BIG;          // MED  y -> p_prev0
    float* bufS1 = bufS0 + MED;         // MED  s_prev
    float* bufT0 = bufS1 + MED;         // SM   m_prev2
    float* bufT1 = bufT0 + SM;          // SM   onehot -> p_next2
    // pre-swizzled f16 weight windows (each 8*16*32*16 = 65536 f16 = 128 KB)
    const size_t WWIN = 65536;
    _Float16* wbActA = (_Float16*)(bufT1 + SM);
    _Float16* wbActB = wbActA + WWIN;
    _Float16* wbIhA  = wbActB + WWIN;
    _Float16* wbIhB  = wbIhA + WWIN;
    _Float16* wbHh   = wbIhB + WWIN;
    _Float16* wbUn   = wbHh + WWIN;
    _Float16* wbUp   = wbUn + WWIN;

    const int gH = NHID / 128;                  // 512 blocks for M=65536
    const int gI = NIN / 128;                   // 128 blocks for M=16384
    const int gO = (NOUT + 127) / 128;          // 8 blocks for M=1000
    const int zB = (int)((BIG + 255) / 256);

    // ---- one-time weight swizzles (tiny) ----
    wswz_k<<<16, 256, 0, stream>>>(Wact,       512, wbActA);
    wswz_k<<<16, 256, 0, stream>>>(Wact + 256, 512, wbActB);
    wswz_k<<<16, 256, 0, stream>>>(Wih,        512, wbIhA);
    wswz_k<<<16, 256, 0, stream>>>(Wih + 256,  512, wbIhB);
    wswz_k<<<16, 256, 0, stream>>>(Whh,        256, wbHh);
    wswz_k<<<16, 256, 0, stream>>>(Wun,        256, wbUn);
    wswz_k<<<16, 256, 0, stream>>>(Wup,        256, wbUp);

    // ---- forward sweep ----
    outer_k<<<NIN, 256, 0, stream>>>(x, Wvec, bvec, bufS0);                       // y
    spmm_k<<<NHID, 256, 0, stream>>>(cols0, val0, bufS0, bufA, K0);               // m_prev0
    gemm_wmma_k<2, 1><<<gH, 256, LDSB, stream>>>(bufA, h0, nullptr,
        wbActA, wbActB, nullptr, bact, nullptr, bufB, NHID);                      // a0
    spmm_k<<<NHID, 256, 0, stream>>>(cols1, val1, bufB, bufC, K1);                // m_prev1
    gemm_wmma_k<2, 1><<<gH, 256, LDSB, stream>>>(bufC, h1, nullptr,
        wbActA, wbActB, nullptr, bact, nullptr, bufB, NHID);                      // a1
    spmm_k<<<NOUT, 256, 0, stream>>>(cols2, val2, bufB, bufT0, K2);               // m_prev2

    // ---- backward RNN sweep ----
    onehot_k<<<NOUT, 256, 0, stream>>>(Yp, bufT1);                                // m_next (one-hot)
    gemm_wmma_k<3, 2><<<gO, 256, LDSB, stream>>>(bufT1, bufT0, h2,
        wbIhA, wbIhB, wbHh, bih, bhh, nh2, NOUT);                                 // new_h2

    zero_k<<<zB, 256, 0, stream>>>(bufB, BIG);
    spmm_t_k<<<(nnz2 + 7) / 8, 256, 0, stream>>>(rows2, cols2, val2, nh2, bufB, nnz2); // m_next1
    gemm_wmma_k<3, 2><<<gH, 256, LDSB, stream>>>(bufB, bufC, h1,
        wbIhA, wbIhB, wbHh, bih, bhh, nh1, NHID);                                 // new_h1

    zero_k<<<zB, 256, 0, stream>>>(bufB, BIG);
    spmm_t_k<<<(nnz1 + 7) / 8, 256, 0, stream>>>(rows1, cols1, val1, nh1, bufB, nnz1); // m_next0
    gemm_wmma_k<3, 2><<<gH, 256, LDSB, stream>>>(bufB, bufA, h0,
        wbIhA, wbIhB, wbHh, bih, bhh, nh0, NHID);                                 // new_h0
    // (spmm_t for j=0 is dead code in the reference: result unused)

    // ---- plasticity updates ----
    outer_k<<<NIN, 256, 0, stream>>>(x, Wfirst, bfirst, bufS1);                   // s_prev
    gemm_wmma_k<1, 3><<<gI, 256, LDSB, stream>>>(bufS1, nullptr, nullptr,
        wbUp, nullptr, nullptr, bup, nullptr, bufS0, NIN);                        // p_prev0
    gemm_wmma_k<1, 3><<<gH, 256, LDSB, stream>>>(nh0, nullptr, nullptr,
        wbUn, nullptr, nullptr, bun, nullptr, bufC, NHID);                        // p_next0
    dw_k<<<(nnz0 + 7) / 8, 256, 0, stream>>>(rows0, cols0, val0, bufS0, bufC, nv0, nnz0);

    gemm_wmma_k<1, 3><<<gH, 256, LDSB, stream>>>(nh0, nullptr, nullptr,
        wbUp, nullptr, nullptr, bup, nullptr, bufA, NHID);                        // p_prev1
    gemm_wmma_k<1, 3><<<gH, 256, LDSB, stream>>>(nh1, nullptr, nullptr,
        wbUn, nullptr, nullptr, bun, nullptr, bufC, NHID);                        // p_next1
    dw_k<<<(nnz1 + 7) / 8, 256, 0, stream>>>(rows1, cols1, val1, bufA, bufC, nv1, nnz1);

    gemm_wmma_k<1, 3><<<gH, 256, LDSB, stream>>>(nh1, nullptr, nullptr,
        wbUp, nullptr, nullptr, bup, nullptr, bufB, NHID);                        // p_prev2
    gemm_wmma_k<1, 3><<<gO, 256, LDSB, stream>>>(nh2, nullptr, nullptr,
        wbUn, nullptr, nullptr, bun, nullptr, bufT1, NOUT);                       // p_next2
    dw_k<<<(nnz2 + 7) / 8, 256, 0, stream>>>(rows2, cols2, val2, bufB, bufT1, nv2, nnz2);
}
